// MultiHeadDotProduct_67087389163659
// MI455X (gfx1250) — matrix operand order repaired and checked by reading, not compiled, not run on hardware
//
#include <hip/hip_runtime.h>
#include <math.h>

#define N_NODES 8192
#define DEG 32
#define EMBED 256
#define NHEAD 8
#define HDIM 32

typedef float     v8f  __attribute__((ext_vector_type(8)));
typedef _Float16  v8h  __attribute__((ext_vector_type(8)));
typedef _Float16  v16h __attribute__((ext_vector_type(16)));

// ---------------------------------------------------------------------------
// f32 -> f16 conversion (element counts here are always multiples of 4)
// ---------------------------------------------------------------------------
__global__ __launch_bounds__(256) void cvt_f32_f16(
    const float* __restrict__ in, _Float16* __restrict__ out, int n)
{
  const int i = (blockIdx.x * blockDim.x + threadIdx.x) * 4;
  if (i + 3 < n) {
    float4 v = *(const float4*)(in + i);
    out[i + 0] = (_Float16)v.x;
    out[i + 1] = (_Float16)v.y;
    out[i + 2] = (_Float16)v.z;
    out[i + 3] = (_Float16)v.w;
  }
}

// ---------------------------------------------------------------------------
// Y = X * W^T + bias, f16 inputs, f32 accumulate via v_wmma_f32_16x16x32_f16.
// X: rows x 256 (f16), W: 256 x 256 row-major [out][in] (f16).
// One wave32 computes a 16(M) x 64(N) tile: 1 A fragment reused across 4 B
// fragments / 4 accumulators per K-step (8 K-steps of 32).
//
// A frag (16x32 f16): lane<16 -> row=lane,    halves K+0..15
//                     lane>=16 -> row=lane-16, halves K+16..31
// B frag (32x16 f16): lane<16 -> col=lane,    halves K+0..15 (contig in W row)
//                     lane>=16 -> col=lane-16, halves K+16..31
// C/D: VGPR j -> row m0 + j + 8*(lane>=16), col = tile base + (lane&15)
// ---------------------------------------------------------------------------
template <typename OutT>
__global__ __launch_bounds__(256) void gemm_f16_wmma(
    const _Float16* __restrict__ X, const _Float16* __restrict__ W,
    const float* __restrict__ bias, OutT* __restrict__ Y, int rows)
{
  const int lane = threadIdx.x & 31;
  const int wave = threadIdx.x >> 5;
  const int tile = blockIdx.x * (blockDim.x >> 5) + wave;
  const int colGroups = EMBED / 64;              // 4 groups of 64 cols
  const int m0 = (tile / colGroups) * 16;
  const int o0 = (tile % colGroups) * 64;
  if (m0 >= rows) return;

  const int r  = lane & 15;
  const int hi = lane >> 4;                      // 0 or 1

  const _Float16* arow = X + (size_t)(m0 + r) * EMBED + hi * 16;
  const _Float16* b0   = W + (size_t)(o0 +  0 + r) * EMBED + hi * 16;
  const _Float16* b1   = W + (size_t)(o0 + 16 + r) * EMBED + hi * 16;
  const _Float16* b2   = W + (size_t)(o0 + 32 + r) * EMBED + hi * 16;
  const _Float16* b3   = W + (size_t)(o0 + 48 + r) * EMBED + hi * 16;

  v8f c0, c1, c2, c3;
  #pragma unroll
  for (int j = 0; j < 8; ++j) {
    c0[j] = bias[o0 +  0 + r];
    c1[j] = bias[o0 + 16 + r];
    c2[j] = bias[o0 + 32 + r];
    c3[j] = bias[o0 + 48 + r];
  }

  #pragma unroll
  for (int k = 0; k < EMBED; k += 32) {
    v8h  alo = *(const v8h*)(arow + k);
    v8h  ahi = *(const v8h*)(arow + k + 8);
    v16h a   = __builtin_shufflevector(alo, ahi,
                 0,1,2,3,4,5,6,7,8,9,10,11,12,13,14,15);
    v16h bb0 = *(const v16h*)(b0 + k);
    v16h bb1 = *(const v16h*)(b1 + k);
    v16h bb2 = *(const v16h*)(b2 + k);
    v16h bb3 = *(const v16h*)(b3 + k);
    // (neg_a, A, neg_b, B, c_mod, C, reuse_a, reuse_b)
    c0 = __builtin_amdgcn_wmma_f32_16x16x32_f16(false, a, false, bb0,
                                                (short)0, c0, false, false);
    c1 = __builtin_amdgcn_wmma_f32_16x16x32_f16(false, a, false, bb1,
                                                (short)0, c1, false, false);
    c2 = __builtin_amdgcn_wmma_f32_16x16x32_f16(false, a, false, bb2,
                                                (short)0, c2, false, false);
    c3 = __builtin_amdgcn_wmma_f32_16x16x32_f16(false, a, false, bb3,
                                                (short)0, c3, false, false);
  }

  OutT* yout = Y + (size_t)(m0 + hi * 8) * EMBED + o0 + r;
  #pragma unroll
  for (int j = 0; j < 8; ++j) {
    yout[(size_t)j * EMBED +  0] = (OutT)c0[j];
    yout[(size_t)j * EMBED + 16] = (OutT)c1[j];
    yout[(size_t)j * EMBED + 32] = (OutT)c2[j];
    yout[(size_t)j * EMBED + 48] = (OutT)c3[j];
  }
}

// ---------------------------------------------------------------------------
// One wave32 per (node, head); f16 Q/K/V in, all math f32, f16 AO out.
// Phase 1: lane = edge slot: sim = dot(q, k[src]) / sqrt(HDIM) (q via shfl).
// Wave softmax via shfl_xor. Phase 2: lane = output dim, weighted V sum.
// ---------------------------------------------------------------------------
__global__ __launch_bounds__(256) void edge_softmax_attn(
    const _Float16* __restrict__ Q, const _Float16* __restrict__ K,
    const _Float16* __restrict__ V, const int* __restrict__ edge_index,
    _Float16* __restrict__ AO)
{
  const int lane = threadIdx.x & 31;
  const int wid  = (blockIdx.x * blockDim.x + threadIdx.x) >> 5;
  const int n = wid >> 3;       // destination node
  const int h = wid & 7;        // head
  if (n >= N_NODES) return;

  const int e0 = n * DEG;
  const int   mySrc = edge_index[(size_t)(e0 + lane) * 2];   // [src, dst] rows
  const float qreg  = (float)Q[(size_t)n * EMBED + h * HDIM + lane];

  const _Float16* krow = K + (size_t)mySrc * EMBED + h * HDIM;
  float sim = 0.f;
  #pragma unroll
  for (int d = 0; d < HDIM; d += 8) {
    v8h kv = *(const v8h*)(krow + d);
    #pragma unroll
    for (int j = 0; j < 8; ++j)
      sim += (float)kv[j] * __shfl(qreg, d + j, 32);
  }
  sim *= 0.17677669529663688f;   // 1/sqrt(32)

  float m = sim;
  #pragma unroll
  for (int off = 16; off > 0; off >>= 1)
    m = fmaxf(m, __shfl_xor(m, off, 32));
  const float p = __expf(sim - m);
  float s = p;
  #pragma unroll
  for (int off = 16; off > 0; off >>= 1)
    s += __shfl_xor(s, off, 32);
  const float attn = p / s;

  float out = 0.f;
  #pragma unroll 4
  for (int a = 0; a < DEG; ++a) {
    const float w  = __shfl(attn,  a, 32);
    const int   sa = __shfl(mySrc, a, 32);
    out += w * (float)V[(size_t)sa * EMBED + h * HDIM + lane];
  }
  AO[(size_t)n * EMBED + h * HDIM + lane] = (_Float16)out;
}

// ---------------------------------------------------------------------------
// d_in: 0 feats, 1 edge_index, 2 edge_attr(unused), 3 Wq, 4 bq, 5 Wk, 6 bk,
//       7 Wv, 8 bv, 9 Wo, 10 bo
// ---------------------------------------------------------------------------
extern "C" void kernel_launch(void* const* d_in, const int* in_sizes, int n_in,
                              void* d_out, int out_size, void* d_ws, size_t ws_size,
                              hipStream_t stream)
{
  const float* feats      = (const float*)d_in[0];
  const int*   edge_index = (const int*)  d_in[1];
  const float* Wq = (const float*)d_in[3];
  const float* bq = (const float*)d_in[4];
  const float* Wk = (const float*)d_in[5];
  const float* bk = (const float*)d_in[6];
  const float* Wv = (const float*)d_in[7];
  const float* bv = (const float*)d_in[8];
  const float* Wo = (const float*)d_in[9];
  const float* bo = (const float*)d_in[10];

  const size_t NF = (size_t)N_NODES * EMBED;   // 2M elems
  const size_t NW = (size_t)EMBED * EMBED;     // 64K elems

  _Float16* XH  = (_Float16*)d_ws;
  _Float16* WqH = XH  + NF;
  _Float16* WkH = WqH + NW;
  _Float16* WvH = WkH + NW;
  _Float16* WoH = WvH + NW;
  _Float16* QH  = WoH + NW;
  _Float16* KH  = QH  + NF;
  _Float16* VH  = KH  + NF;
  _Float16* AOH = VH  + NF;

  const int cvtB_f = (int)(NF / 4 + 255) / 256;
  const int cvtB_w = (int)(NW / 4 + 255) / 256;
  cvt_f32_f16<<<cvtB_f, 256, 0, stream>>>(feats, XH,  (int)NF);
  cvt_f32_f16<<<cvtB_w, 256, 0, stream>>>(Wq,    WqH, (int)NW);
  cvt_f32_f16<<<cvtB_w, 256, 0, stream>>>(Wk,    WkH, (int)NW);
  cvt_f32_f16<<<cvtB_w, 256, 0, stream>>>(Wv,    WvH, (int)NW);
  cvt_f32_f16<<<cvtB_w, 256, 0, stream>>>(Wo,    WoH, (int)NW);

  // 512 row tiles x 4 col groups = 2048 waves; 8 waves per 256-thread block
  const int gemmBlocks = (N_NODES / 16) * (EMBED / 64) / 8;
  gemm_f16_wmma<_Float16><<<gemmBlocks, 256, 0, stream>>>(XH, WqH, bq, QH, N_NODES);
  gemm_f16_wmma<_Float16><<<gemmBlocks, 256, 0, stream>>>(XH, WkH, bk, KH, N_NODES);
  gemm_f16_wmma<_Float16><<<gemmBlocks, 256, 0, stream>>>(XH, WvH, bv, VH, N_NODES);

  const int waves = N_NODES * NHEAD;   // 65536 (node,head) waves
  edge_softmax_attn<<<waves / 8, 256, 0, stream>>>(QH, KH, VH, edge_index, AOH);

  gemm_f16_wmma<float><<<gemmBlocks, 256, 0, stream>>>(AOH, WoH, bo,
                                                       (float*)d_out, N_NODES);
}
